// Particle_82240033783923
// MI455X (gfx1250) — compile-verified, matrix operand order checked by reading, not run on hardware
//
#include <hip/hip_runtime.h>
#include <hip/hip_fp16.h>

#define NP 50000
#define NE 800000
#define SDIM 32
#define OCH 16
#define HDIM 32
#define BN_EPS 1e-5f
#define EGROUPS 6250   // (NE/16 tiles) / 8 waves
#define NGROUPS 6250   // NP / 8 nodes per block

typedef __attribute__((ext_vector_type(16))) _Float16 v16h;
typedef __attribute__((ext_vector_type(8)))  float    v8f;

union V16 { v16h v; __half2 h2[8]; unsigned u[8]; };

#define V8F_ZERO {0.f,0.f,0.f,0.f,0.f,0.f,0.f,0.f}

__device__ __forceinline__ v8f wmma_f16(const V16& a, const V16& b, v8f c) {
  // D = A(16x32 f16) x B(32x16 f16) + C(f32)
  return __builtin_amdgcn_wmma_f32_16x16x32_f16(false, a.v, false, b.v,
                                                (short)0, c, false, false);
}

// Load a 32x16 B-operand tile from a row-major fp32 weight [K x ld], cols n0..n0+15.
// B layout mirrors the documented 16-bit A layout (K packed 2/VGPR, split halves).
__device__ __forceinline__ V16 load_b_tile(const float* w, int ld, int n0) {
  int lane = threadIdx.x & 31;
  int n  = n0 + (lane & 15);
  int kb = (lane >> 4) * 8;
  V16 b;
#pragma unroll
  for (int j = 0; j < 8; ++j) {
    int k = (j < 4) ? (kb + 2 * j) : (16 + kb + 2 * (j - 4));
    b.h2[j] = __floats2half2_rn(w[k * ld + n], w[(k + 1) * ld + n]);
  }
  return b;
}

// Load a 16x32 A-operand tile from an LDS f16 row-major [16][32] buffer.
__device__ __forceinline__ V16 load_a_lds(const __half* buf) {
  int lane = threadIdx.x & 31;
  int m  = lane & 15;
  int kb = (lane >> 4) * 8;
  V16 a;
  uint4 q0 = *reinterpret_cast<const uint4*>(buf + m * 32 + kb);
  uint4 q1 = *reinterpret_cast<const uint4*>(buf + m * 32 + 16 + kb);
  a.u[0]=q0.x; a.u[1]=q0.y; a.u[2]=q0.z; a.u[3]=q0.w;
  a.u[4]=q1.x; a.u[5]=q1.y; a.u[6]=q1.z; a.u[7]=q1.w;
  return a;
}

// 16 contiguous fp32 -> 16 f16 into LDS (two b128 stores).
__device__ __forceinline__ void stage16(__half* dst, const float* src) {
  const float4* s = reinterpret_cast<const float4*>(src);
  float4 f0 = s[0], f1 = s[1], f2 = s[2], f3 = s[3];
  union { uint4 q; __half2 h[4]; } u0, u1;
  u0.h[0]=__floats2half2_rn(f0.x,f0.y); u0.h[1]=__floats2half2_rn(f0.z,f0.w);
  u0.h[2]=__floats2half2_rn(f1.x,f1.y); u0.h[3]=__floats2half2_rn(f1.z,f1.w);
  u1.h[0]=__floats2half2_rn(f2.x,f2.y); u1.h[1]=__floats2half2_rn(f2.z,f2.w);
  u1.h[2]=__floats2half2_rn(f3.x,f3.y); u1.h[3]=__floats2half2_rn(f3.z,f3.w);
  reinterpret_cast<uint4*>(dst)[0] = u0.q;
  reinterpret_cast<uint4*>(dst)[1] = u1.q;
}

// Apply bias + BN + ReLU to two 16x16 f32 accumulators and store as f16 tile [16][32].
__device__ __forceinline__ void acc_store_h16(__half* buf, v8f c0, v8f c1,
    float bb0, float bb1, float sc0, float sc1, float sh0, float sh1) {
  int lane = threadIdx.x & 31;
  int n  = lane & 15;
  int rb = (lane >> 4) * 8;
#pragma unroll
  for (int j = 0; j < 8; ++j) {
    float a = fmaxf((c0[j] + bb0) * sc0 + sh0, 0.f);
    float b = fmaxf((c1[j] + bb1) * sc1 + sh1, 0.f);
    buf[(rb + j) * 32 + n]      = __float2half(a);
    buf[(rb + j) * 32 + 16 + n] = __float2half(b);
  }
}

// ---------------- BN finalize: scale = g*rsqrt(var+eps), shift = be - mean*scale
__global__ void kfin(const float* stats, const float* gamma, const float* beta,
                     float M, float* bn) {
  int t = threadIdx.x;
  if (t < 32) {
    float mean = stats[t] / M;
    float var  = stats[32 + t] / M - mean * mean;
    float S    = gamma[t] * rsqrtf(var + BN_EPS);
    bn[t]      = S;
    bn[32 + t] = beta[t] - mean * S;
  }
}

// ---------------- Edge stats pass 1: column sum/sumsq of a1 = ea@w1 + b1
__global__ __launch_bounds__(256) void ke_stats1(const float* __restrict__ ea,
    const float* __restrict__ w1, const float* __restrict__ b1, float* stats) {
  __shared__ __align__(16) __half sbuf[8][512];
  __shared__ float redS[32], redQ[32];
  int tid = threadIdx.x, wave = tid >> 5, lane = tid & 31;
  int n = lane & 15, p = lane >> 4, m = lane & 15;
  V16 B0 = load_b_tile(w1, 32, 0), B1 = load_b_tile(w1, 32, 16);
  float bb0 = b1[n], bb1 = b1[16 + n];
  float s0 = 0, q0 = 0, s1 = 0, q1 = 0;
  for (int g = blockIdx.x; g < EGROUPS; g += gridDim.x) {
    int e0 = (g * 8 + wave) * 16;
    stage16(&sbuf[wave][m * 32 + p * 16], ea + (size_t)(e0 + m) * SDIM + p * 16);
    __syncthreads();
    V16 A = load_a_lds(sbuf[wave]);
    v8f c0 = V8F_ZERO, c1 = V8F_ZERO;
    c0 = wmma_f16(A, B0, c0);
    c1 = wmma_f16(A, B1, c1);
#pragma unroll
    for (int j = 0; j < 8; ++j) {
      float a = c0[j] + bb0; s0 += a; q0 += a * a;
      float b = c1[j] + bb1; s1 += b; q1 += b * b;
    }
    __syncthreads();
  }
  if (tid < 32) { redS[tid] = 0.f; redQ[tid] = 0.f; }
  __syncthreads();
  atomicAdd(&redS[n], s0);      atomicAdd(&redQ[n], q0);
  atomicAdd(&redS[16 + n], s1); atomicAdd(&redQ[16 + n], q1);
  __syncthreads();
  if (tid < 32) { atomicAdd(&stats[tid], redS[tid]); atomicAdd(&stats[32 + tid], redQ[tid]); }
}

// ---------------- Edge stats pass 2: stats of a2 = relu(bn1(a1))@w2 + b2
__global__ __launch_bounds__(256) void ke_stats2(const float* __restrict__ ea,
    const float* __restrict__ w1, const float* __restrict__ b1,
    const float* __restrict__ w2, const float* __restrict__ b2,
    const float* __restrict__ bn1, float* stats) {
  __shared__ __align__(16) __half sbuf[8][512];
  __shared__ float redS[32], redQ[32];
  int tid = threadIdx.x, wave = tid >> 5, lane = tid & 31;
  int n = lane & 15, p = lane >> 4, m = lane & 15;
  V16 B10 = load_b_tile(w1, 32, 0), B11 = load_b_tile(w1, 32, 16);
  V16 B20 = load_b_tile(w2, 32, 0), B21 = load_b_tile(w2, 32, 16);
  float bb10 = b1[n], bb11 = b1[16 + n], bb20 = b2[n], bb21 = b2[16 + n];
  float sc0 = bn1[n], sc1 = bn1[16 + n], sh0 = bn1[32 + n], sh1 = bn1[48 + n];
  float s0 = 0, q0 = 0, s1 = 0, q1 = 0;
  for (int g = blockIdx.x; g < EGROUPS; g += gridDim.x) {
    int e0 = (g * 8 + wave) * 16;
    stage16(&sbuf[wave][m * 32 + p * 16], ea + (size_t)(e0 + m) * SDIM + p * 16);
    __syncthreads();
    V16 A = load_a_lds(sbuf[wave]);
    __syncthreads();
    v8f c0 = V8F_ZERO, c1 = V8F_ZERO;
    c0 = wmma_f16(A, B10, c0);
    c1 = wmma_f16(A, B11, c1);
    acc_store_h16(sbuf[wave], c0, c1, bb10, bb11, sc0, sc1, sh0, sh1); // h1
    __syncthreads();
    V16 A2 = load_a_lds(sbuf[wave]);
    v8f d0 = V8F_ZERO, d1 = V8F_ZERO;
    d0 = wmma_f16(A2, B20, d0);
    d1 = wmma_f16(A2, B21, d1);
#pragma unroll
    for (int j = 0; j < 8; ++j) {
      float a = d0[j] + bb20; s0 += a; q0 += a * a;
      float b = d1[j] + bb21; s1 += b; q1 += b * b;
    }
    __syncthreads();
  }
  if (tid < 32) { redS[tid] = 0.f; redQ[tid] = 0.f; }
  __syncthreads();
  atomicAdd(&redS[n], s0);      atomicAdd(&redQ[n], q0);
  atomicAdd(&redS[16 + n], s1); atomicAdd(&redQ[16 + n], q1);
  __syncthreads();
  if (tid < 32) { atomicAdd(&stats[tid], redS[tid]); atomicAdd(&stats[32 + tid], redQ[tid]); }
}

// ---------------- Heavy kernel: layers 1-3 + bilinear message + scatter-add
__global__ __launch_bounds__(256) void ke_main(const float* __restrict__ ea,
    const float* __restrict__ x, const int* __restrict__ srcI, const int* __restrict__ dstI,
    const float* __restrict__ w1, const float* __restrict__ b1,
    const float* __restrict__ w2, const float* __restrict__ b2,
    const float* __restrict__ w3, const float* __restrict__ b3,
    const float* __restrict__ bn1, const float* __restrict__ bn2,
    float* __restrict__ agg) {
  __shared__ __align__(16) __half w3T[16 * 1024];  // [n][kk] transposed f16, 32 KB
  __shared__ __align__(16) __half bufA[8][512];    // ea -> h1 -> h2, per-wave
  __shared__ __align__(16) __half bufB[8][512];    // x[src] f16, per-wave
  int tid = threadIdx.x, wave = tid >> 5, lane = tid & 31;
  int n = lane & 15, p = lane >> 4, m = lane & 15;
  int kb = (lane >> 4) * 8, rb = (lane >> 4) * 8;

  // Stage w3 (flat idx = kk*16 + n) transposed into LDS as f16
  for (int i = tid; i < 16384; i += 256) {
    int col = i & 15, kk = i >> 4;
    w3T[col * 1024 + kk] = __float2half(w3[i]);
  }
  V16 B10 = load_b_tile(w1, 32, 0), B11 = load_b_tile(w1, 32, 16);
  V16 B20 = load_b_tile(w2, 32, 0), B21 = load_b_tile(w2, 32, 16);
  V16 Bb3 = load_b_tile(b3, 16, 0);  // bias "chunk": B[i][o] = b3[i*16+o]
  float bb10 = b1[n], bb11 = b1[16 + n], bb20 = b2[n], bb21 = b2[16 + n];
  float s10 = bn1[n], s11 = bn1[16 + n], h10 = bn1[32 + n], h11 = bn1[48 + n];
  float s20 = bn2[n], s21 = bn2[16 + n], h20 = bn2[32 + n], h21 = bn2[48 + n];

  int tile = blockIdx.x * 8 + wave;
  int e0 = tile * 16;

  // Stage edge_attr tile and gathered x[src] tile (f16)
  stage16(&bufA[wave][m * 32 + p * 16], ea + (size_t)(e0 + m) * SDIM + p * 16);
  int sidx = srcI[e0 + m];
  stage16(&bufB[wave][m * 32 + p * 16], x + (size_t)sidx * SDIM + p * 16);
  __syncthreads();

  V16 Aea = load_a_lds(bufA[wave]);
  V16 Axs = load_a_lds(bufB[wave]);
  __syncthreads();
  v8f c0 = V8F_ZERO, c1 = V8F_ZERO;
  c0 = wmma_f16(Aea, B10, c0);
  c1 = wmma_f16(Aea, B11, c1);
  acc_store_h16(bufA[wave], c0, c1, bb10, bb11, s10, s11, h10, h11);  // h1
  __syncthreads();
  V16 A1 = load_a_lds(bufA[wave]);
  __syncthreads();
  v8f d0 = V8F_ZERO, d1 = V8F_ZERO;
  d0 = wmma_f16(A1, B20, d0);
  d1 = wmma_f16(A1, B21, d1);
  acc_store_h16(bufA[wave], d0, d1, bb20, bb21, s20, s21, h20, h21);  // h2
  __syncthreads();

  // msg = x_src @ (h2 @ w3 reshaped) + x_src @ b3  — K=1024 bilinear via 33 WMMAs
  v8f macc = V8F_ZERO;
  macc = wmma_f16(Axs, Bb3, macc);  // bias chunk
  const __half* h2row = &bufA[wave][m * 32];
  const __half* bw = w3T + n * 1024;
#pragma unroll 4
  for (int c = 0; c < 32; ++c) {
    __half2 hd = __half2half2(h2row[c]);  // h2[m][c] broadcast
    V16 Ac;
#pragma unroll
    for (int j = 0; j < 8; ++j) Ac.h2[j] = __hmul2(Axs.h2[j], hd);
    V16 Bc;
    uint4 q0 = *reinterpret_cast<const uint4*>(bw + c * 32 + kb);
    uint4 q1 = *reinterpret_cast<const uint4*>(bw + c * 32 + 16 + kb);
    Bc.u[0]=q0.x; Bc.u[1]=q0.y; Bc.u[2]=q0.z; Bc.u[3]=q0.w;
    Bc.u[4]=q1.x; Bc.u[5]=q1.y; Bc.u[6]=q1.z; Bc.u[7]=q1.w;
    macc = wmma_f16(Ac, Bc, macc);
  }

  // Scatter-add into agg[N,16]
#pragma unroll
  for (int j = 0; j < 8; ++j) {
    int row = rb + j;
    int d = dstI[e0 + row];
    atomicAdd(&agg[(size_t)d * OCH + n], macc[j]);
  }
}

// ---------------- messages = agg + x @ root_w + root_b
__global__ __launch_bounds__(256) void kn_messages(const float* __restrict__ x,
    const float* __restrict__ agg, const float* __restrict__ rw,
    const float* __restrict__ rb, float* __restrict__ msgs) {
  int t = blockIdx.x * 256 + threadIdx.x;
  if (t >= NP * OCH) return;
  int node = t >> 4, o = t & 15;
  float a = agg[t] + rb[o];
  const float* xr = x + (size_t)node * SDIM;
#pragma unroll
  for (int i = 0; i < 32; ++i) a += xr[i] * rw[i * 16 + o];
  msgs[t] = a;
}

// ---------------- node stats pass 1: o1 = [x,msg]@own_w1 + b1
__global__ __launch_bounds__(256) void kn_stats1(const float* __restrict__ x,
    const float* __restrict__ msgs, const float* __restrict__ w1,
    const float* __restrict__ b1, float* stats) {
  __shared__ float w1s[48 * 32];
  __shared__ float comb[8][48];
  __shared__ float redS[32], redQ[32];
  int tid = threadIdx.x, nl = tid >> 5, o = tid & 31;
  for (int i = tid; i < 48 * 32; i += 256) w1s[i] = w1[i];
  float s = 0, q = 0;
  __syncthreads();
  for (int g = blockIdx.x; g < NGROUPS; g += gridDim.x) {
    int base = g * 8;
    for (int i = tid; i < 384; i += 256) {
      int r = i / 48, c = i % 48, node = base + r;
      comb[r][c] = (c < 32) ? x[(size_t)node * SDIM + c]
                            : msgs[(size_t)node * OCH + (c - 32)];
    }
    __syncthreads();
    float a = b1[o];
#pragma unroll
    for (int i = 0; i < 48; ++i) a += comb[nl][i] * w1s[i * 32 + o];
    s += a; q += a * a;
    __syncthreads();
  }
  if (tid < 32) { redS[tid] = 0.f; redQ[tid] = 0.f; }
  __syncthreads();
  atomicAdd(&redS[o], s); atomicAdd(&redQ[o], q);
  __syncthreads();
  if (tid < 32) { atomicAdd(&stats[tid], redS[tid]); atomicAdd(&stats[32 + tid], redQ[tid]); }
}

// ---------------- node stats pass 2
__global__ __launch_bounds__(256) void kn_stats2(const float* __restrict__ x,
    const float* __restrict__ msgs, const float* __restrict__ w1,
    const float* __restrict__ b1, const float* __restrict__ w2,
    const float* __restrict__ b2, const float* __restrict__ bn1, float* stats) {
  __shared__ float w1s[48 * 32];
  __shared__ float w2s[32 * 32];
  __shared__ float comb[8][48];
  __shared__ float p1[8][32];
  __shared__ float redS[32], redQ[32];
  int tid = threadIdx.x, nl = tid >> 5, o = tid & 31;
  for (int i = tid; i < 48 * 32; i += 256) w1s[i] = w1[i];
  for (int i = tid; i < 32 * 32; i += 256) w2s[i] = w2[i];
  float S1 = bn1[o], H1 = bn1[32 + o];
  float s = 0, q = 0;
  __syncthreads();
  for (int g = blockIdx.x; g < NGROUPS; g += gridDim.x) {
    int base = g * 8;
    for (int i = tid; i < 384; i += 256) {
      int r = i / 48, c = i % 48, node = base + r;
      comb[r][c] = (c < 32) ? x[(size_t)node * SDIM + c]
                            : msgs[(size_t)node * OCH + (c - 32)];
    }
    __syncthreads();
    float a = b1[o];
#pragma unroll
    for (int i = 0; i < 48; ++i) a += comb[nl][i] * w1s[i * 32 + o];
    p1[nl][o] = fmaxf(a * S1 + H1, 0.f);
    __syncthreads();
    float c2 = b2[o];
#pragma unroll
    for (int h = 0; h < 32; ++h) c2 += p1[nl][h] * w2s[h * 32 + o];
    s += c2; q += c2 * c2;
    __syncthreads();
  }
  if (tid < 32) { redS[tid] = 0.f; redQ[tid] = 0.f; }
  __syncthreads();
  atomicAdd(&redS[o], s); atomicAdd(&redQ[o], q);
  __syncthreads();
  if (tid < 32) { atomicAdd(&stats[tid], redS[tid]); atomicAdd(&stats[32 + tid], redQ[tid]); }
}

// ---------------- node final: new_state
__global__ __launch_bounds__(256) void kn_final(const float* __restrict__ x,
    const float* __restrict__ msgs, const float* __restrict__ w1,
    const float* __restrict__ b1, const float* __restrict__ w2,
    const float* __restrict__ b2, const float* __restrict__ w3,
    const float* __restrict__ b3, const float* __restrict__ bn1,
    const float* __restrict__ bn2, float* __restrict__ out) {
  __shared__ float w1s[48 * 32];
  __shared__ float w2s[32 * 32];
  __shared__ float w3s[32 * 32];
  __shared__ float comb[8][48];
  __shared__ float p1[8][32];
  __shared__ float p2[8][32];
  int tid = threadIdx.x, nl = tid >> 5, o = tid & 31;
  for (int i = tid; i < 48 * 32; i += 256) w1s[i] = w1[i];
  for (int i = tid; i < 32 * 32; i += 256) w2s[i] = w2[i];
  for (int i = tid; i < 32 * 32; i += 256) w3s[i] = w3[i];
  float S1 = bn1[o], H1 = bn1[32 + o], S2 = bn2[o], H2 = bn2[32 + o];
  int base = blockIdx.x * 8;
  for (int i = tid; i < 384; i += 256) {
    int r = i / 48, c = i % 48, node = base + r;
    comb[r][c] = (c < 32) ? x[(size_t)node * SDIM + c]
                          : msgs[(size_t)node * OCH + (c - 32)];
  }
  __syncthreads();
  float a = b1[o];
#pragma unroll
  for (int i = 0; i < 48; ++i) a += comb[nl][i] * w1s[i * 32 + o];
  p1[nl][o] = fmaxf(a * S1 + H1, 0.f);
  __syncthreads();
  float c2 = b2[o];
#pragma unroll
  for (int h = 0; h < 32; ++h) c2 += p1[nl][h] * w2s[h * 32 + o];
  p2[nl][o] = fmaxf(c2 * S2 + H2, 0.f);
  __syncthreads();
  float r3 = b3[o];
#pragma unroll
  for (int h = 0; h < 32; ++h) r3 += p2[nl][h] * w3s[h * 32 + o];
  out[(size_t)(base + nl) * SDIM + o] = r3;
}

extern "C" void kernel_launch(void* const* d_in, const int* in_sizes, int n_in,
                              void* d_out, int out_size, void* d_ws, size_t ws_size,
                              hipStream_t stream) {
  const float* x       = (const float*)d_in[0];
  const float* ea      = (const float*)d_in[1];
  const int*   ei      = (const int*)d_in[2];
  const float* msg_w1  = (const float*)d_in[3];
  const float* msg_b1  = (const float*)d_in[4];
  const float* msg_g1  = (const float*)d_in[5];
  const float* msg_be1 = (const float*)d_in[6];
  const float* msg_w2  = (const float*)d_in[7];
  const float* msg_b2  = (const float*)d_in[8];
  const float* msg_g2  = (const float*)d_in[9];
  const float* msg_be2 = (const float*)d_in[10];
  const float* msg_w3  = (const float*)d_in[11];
  const float* msg_b3  = (const float*)d_in[12];
  const float* root_w  = (const float*)d_in[13];
  const float* root_b  = (const float*)d_in[14];
  const float* own_w1  = (const float*)d_in[15];
  const float* own_b1  = (const float*)d_in[16];
  const float* own_g1  = (const float*)d_in[17];
  const float* own_be1 = (const float*)d_in[18];
  const float* own_w2  = (const float*)d_in[19];
  const float* own_b2  = (const float*)d_in[20];
  const float* own_g2  = (const float*)d_in[21];
  const float* own_be2 = (const float*)d_in[22];
  const float* own_w3  = (const float*)d_in[23];
  const float* own_b3  = (const float*)d_in[24];

  float* ws    = (float*)d_ws;
  float* st_e1 = ws + 0;
  float* st_e2 = ws + 64;
  float* st_o1 = ws + 128;
  float* st_o2 = ws + 192;
  float* bn_e1 = ws + 256;
  float* bn_e2 = ws + 320;
  float* bn_o1 = ws + 384;
  float* bn_o2 = ws + 448;
  float* agg   = ws + 512;                 // [NP * OCH]
  float* msgs  = agg + (size_t)NP * OCH;   // [NP * OCH]
  float* out   = (float*)d_out;

  // zero stats + agg (messages fully overwritten)
  hipMemsetAsync(d_ws, 0, (512 + (size_t)NP * OCH) * sizeof(float), stream);

  ke_stats1<<<1250, 256, 0, stream>>>(ea, msg_w1, msg_b1, st_e1);
  kfin<<<1, 32, 0, stream>>>(st_e1, msg_g1, msg_be1, (float)NE, bn_e1);
  ke_stats2<<<1250, 256, 0, stream>>>(ea, msg_w1, msg_b1, msg_w2, msg_b2, bn_e1, st_e2);
  kfin<<<1, 32, 0, stream>>>(st_e2, msg_g2, msg_be2, (float)NE, bn_e2);
  ke_main<<<EGROUPS, 256, 0, stream>>>(ea, x, ei, ei + NE,
      msg_w1, msg_b1, msg_w2, msg_b2, msg_w3, msg_b3, bn_e1, bn_e2, agg);
  kn_messages<<<(NP * OCH + 255) / 256, 256, 0, stream>>>(x, agg, root_w, root_b, msgs);
  kn_stats1<<<625, 256, 0, stream>>>(x, msgs, own_w1, own_b1, st_o1);
  kfin<<<1, 32, 0, stream>>>(st_o1, own_g1, own_be1, (float)NP, bn_o1);
  kn_stats2<<<625, 256, 0, stream>>>(x, msgs, own_w1, own_b1, own_w2, own_b2, bn_o1, st_o2);
  kfin<<<1, 32, 0, stream>>>(st_o2, own_g2, own_be2, (float)NP, bn_o2);
  kn_final<<<NGROUPS, 256, 0, stream>>>(x, msgs, own_w1, own_b1, own_w2, own_b2,
      own_w3, own_b3, bn_o1, bn_o2, out);
}
